// MultiHeadAttention_42623255445732
// MI455X (gfx1250) — matrix operand order repaired and checked by reading, not compiled, not run on hardware
//
#include <hip/hip_runtime.h>
#include <hip/hip_bf16.h>

typedef _Float16 v4h  __attribute__((ext_vector_type(4)));
typedef _Float16 v8h  __attribute__((ext_vector_type(8)));
typedef _Float16 v16h __attribute__((ext_vector_type(16)));
typedef float    v4f  __attribute__((ext_vector_type(4)));
typedef float    v8f  __attribute__((ext_vector_type(8)));

#define BQ  2
#define SQ  2048
#define DQ  1024
#define HQ  16
#define HDQ 64

#if __has_builtin(__builtin_amdgcn_exp2f)
#define EXP2F(x) __builtin_amdgcn_exp2f(x)
#else
#define EXP2F(x) exp2f(x)
#endif

// log2(e) folded into Q so softmax numerators are exp2(score*log2e) == exp(score)
#define QSCALE (0.125f * 1.44269504088896340736f)
// fixed exp2-domain shift (cancels in normalization); keeps f16 P in range
#define PSHIFT (-11.5f)

// ---------------- WMMA fragment helpers (layouts per CDNA5 ISA 7.12.2) ------

// A-frag 16x32 f16: lane holds row (l&15); VGPR0..3 <- K = 8*half + 0..7,
// VGPR4..7 <- K = 16 + 8*half + 0..7. rowp points at (row, k0).
static __device__ inline v16h ld_a_frag(const _Float16* rowp, int half) {
  v8h c0 = *(const v8h*)(rowp + 8 * half);
  v8h c1 = *(const v8h*)(rowp + 16 + 8 * half);
  return __builtin_shufflevector(c0, c1, 0,1,2,3,4,5,6,7,8,9,10,11,12,13,14,15);
}

// B-frag 32x16 f16: lane holds col (l&15); K = 8*half + 0..7, then 16 + 8*half + 0..7.
static __device__ inline v16h ld_b_frag(const _Float16* rowp, int half) {
  v8h c0 = *(const v8h*)(rowp + 8 * half);
  v8h c1 = *(const v8h*)(rowp + 16 + 8 * half);
  return __builtin_shufflevector(c0, c1, 0,1,2,3,4,5,6,7,8,9,10,11,12,13,14,15);
}

static __device__ inline v8f wmma_f16(v16h a, v16h b, v8f c) {
  return __builtin_amdgcn_wmma_f32_16x16x32_f16(false, a, false, b, (short)0, c,
                                                false, false);
}

// ---------------- fp32 -> f16 conversion (vec4) -----------------------------

__global__ void cvt_f32_f16_kernel(const float* __restrict__ in,
                                   _Float16* __restrict__ out, int n4) {
  int i = blockIdx.x * blockDim.x + threadIdx.x;
  if (i < n4) {
    v4f v = ((const v4f*)in)[i];
    ((v4h*)out)[i] = __builtin_convertvector(v, v4h);
  }
}

// ---------------- generic NT GEMM: out[m,n] = oscale*sum_k A[m,k]*Bw[n,k] ---
// mode 0: f16 out, [B,H,S,HD] scatter (Q,K)      (m=(b,s), n=(h,d))
// mode 1: f16 out, [B,H,HD,S] scatter (V^T)
// mode 2: f32 out, row-major [M,D] + bias         (final projection)

static __device__ inline void store_tile(void* out, const float* bias, v8f c,
                                         int mt, int nt, int lane15, int half,
                                         int mode, float oscale) {
  const int n = nt + lane15;
#pragma unroll
  for (int r = 0; r < 8; ++r) {
    const int m = mt + r + 8 * half;
    if (mode == 2) {
      ((float*)out)[(size_t)m * DQ + n] = c[r] * oscale + bias[n];
    } else {
      const int b = m >> 11, s = m & (SQ - 1);
      const int h = n >> 6,  d = n & (HDQ - 1);
      size_t idx;
      if (mode == 0) idx = ((size_t)(b * HQ + h) * SQ + s) * HDQ + d;
      else           idx = ((size_t)(b * HQ + h) * HDQ + d) * SQ + s;
      ((_Float16*)out)[idx] = (_Float16)(c[r] * oscale);
    }
  }
}

__global__ void gemm_nt_kernel(const _Float16* __restrict__ A,
                               const _Float16* __restrict__ Bw,
                               void* __restrict__ out,
                               const float* __restrict__ bias,
                               int K, int mode, float oscale) {
  const int lane   = threadIdx.x & 31;
  const int wave   = threadIdx.x >> 5;
  const int lane15 = lane & 15;
  const int half   = lane >> 4;
  const int m0 = blockIdx.x * 64 + (wave >> 1) * 32;
  const int n0 = blockIdx.y * 64 + (wave & 1) * 32;

  const _Float16* a0p = A + (size_t)(m0 + lane15) * K;
  const _Float16* a1p = a0p + (size_t)16 * K;
  const _Float16* b0p = Bw + (size_t)(n0 + lane15) * K;
  const _Float16* b1p = b0p + (size_t)16 * K;

  v8f c00 = {}, c01 = {}, c10 = {}, c11 = {};
  for (int k0 = 0; k0 < K; k0 += 32) {
    v16h a0 = ld_a_frag(a0p + k0, half);
    v16h a1 = ld_a_frag(a1p + k0, half);
    v16h b0 = ld_b_frag(b0p + k0, half);
    v16h b1 = ld_b_frag(b1p + k0, half);
    c00 = wmma_f16(a0, b0, c00);
    c01 = wmma_f16(a0, b1, c01);
    c10 = wmma_f16(a1, b0, c10);
    c11 = wmma_f16(a1, b1, c11);
  }
  store_tile(out, bias, c00, m0,      n0,      lane15, half, mode, oscale);
  store_tile(out, bias, c01, m0,      n0 + 16, lane15, half, mode, oscale);
  store_tile(out, bias, c10, m0 + 16, n0,      lane15, half, mode, oscale);
  store_tile(out, bias, c11, m0 + 16, n0 + 16, lane15, half, mode, oscale);
}

// ---------------- flash attention: 1 wave = 16 query rows -------------------
// Q (prescaled by 0.125*log2e), K: [B,H,S,HD] f16.  Vt: [B,H,HD,S] f16.
// Streaming softmax without running max (scores bounded: |q.k/8| << 80), with
// row sums produced by an extra ones-column WMMA (o4) instead of lane shuffles.

__global__ void flash_attn_kernel(const _Float16* __restrict__ Q,
                                  const _Float16* __restrict__ Kh,
                                  const _Float16* __restrict__ Vt,
                                  const float* __restrict__ mask,
                                  _Float16* __restrict__ Ctx) {
  __shared__ __align__(16) _Float16 pbuf[4][16 * 32];  // wave-private P tiles
  const int lane   = threadIdx.x & 31;
  const int wave   = threadIdx.x >> 5;
  const int lane15 = lane & 15;
  const int half   = lane >> 4;
  const int h  = blockIdx.y;
  const int b  = blockIdx.z;
  const int bh = b * HQ + h;
  const int q0 = blockIdx.x * 64 + wave * 16;

  // Q fragments for this wave's 16 rows (d = 0..31 and 32..63), kept in regs.
  const _Float16* qrow = Q + ((size_t)bh * SQ + q0 + lane15) * HDQ;
  v16h qa0 = ld_a_frag(qrow, half);
  v16h qa1 = ld_a_frag(qrow + 32, half);

  const _Float16* kbase = Kh + (size_t)bh * SQ * HDQ;
  const _Float16* vlane = Vt + ((size_t)bh * HDQ + lane15) * SQ;  // d = lane15
  const float*    mrow  = mask + (size_t)b * SQ;

  v16h vone;
#pragma unroll
  for (int i = 0; i < 16; ++i) vone[i] = (_Float16)1.0f;

  v8f o0 = {}, o1 = {}, o2 = {}, o3 = {}, o4 = {};

  _Float16* pl = &pbuf[wave][0];
  const _Float16* prow = pl + lane15 * 32;

  for (int j0 = 0; j0 < SQ; j0 += 32) {
    const _Float16* krow0 = kbase + (size_t)(j0 + lane15) * HDQ;
    const _Float16* krow1 = krow0 + 16 * HDQ;
    if (j0 + 32 < SQ)  // hint next K block toward cache
      __builtin_prefetch(kbase + (size_t)(j0 + 32 + lane15) * HDQ, 0, 1);

    // S = Q @ K^T  (two column groups of 16 kv tokens, contract d=64)
    v8f s0 = {}, s1 = {};
    s0 = wmma_f16(qa0, ld_b_frag(krow0, half), s0);
    s0 = wmma_f16(qa1, ld_b_frag(krow0 + 32, half), s0);
    s1 = wmma_f16(qa0, ld_b_frag(krow1, half), s1);
    s1 = wmma_f16(qa1, ld_b_frag(krow1 + 32, half), s1);

    // column mask folded into one additive bias (exp2 domain)
    const float mb0 = (mrow[j0 + lane15]      == 0.0f) ? -1e30f : PSHIFT;
    const float mb1 = (mrow[j0 + 16 + lane15] == 0.0f) ? -1e30f : PSHIFT;

    // P = exp2(S + bias): 1 add + 1 v_exp + cvt + ds_store per element
#pragma unroll
    for (int r = 0; r < 8; ++r) {
      const float p0 = EXP2F(s0[r] + mb0);
      const float p1 = EXP2F(s1[r] + mb1);
      pl[(r + 8 * half) * 32 + lane15]      = (_Float16)p0;
      pl[(r + 8 * half) * 32 + 16 + lane15] = (_Float16)p1;
    }

    // same-wave DS ordering; make P stores visible before the A-frag read
    asm volatile("s_wait_dscnt 0" ::: "memory");

    // O += P(16x32) @ V(32x16) for four d-blocks; o4 += P @ ones (row sums)
    v16h pa = ld_a_frag(prow, half);
    const _Float16* vrow = vlane + j0;
    o0 = wmma_f16(pa, ld_b_frag(vrow, half), o0);
    o1 = wmma_f16(pa, ld_b_frag(vrow + (size_t)16 * SQ, half), o1);
    o2 = wmma_f16(pa, ld_b_frag(vrow + (size_t)32 * SQ, half), o2);
    o3 = wmma_f16(pa, ld_b_frag(vrow + (size_t)48 * SQ, half), o3);
    o4 = wmma_f16(pa, vone, o4);
  }

  // finalize: divide by row sums, write ctx[b, s, h*64 + d] as f16
#pragma unroll
  for (int r = 0; r < 8; ++r) {
    const float inv = 1.0f / o4[r];
    const int srow = q0 + r + 8 * half;
    _Float16* crow = Ctx + ((size_t)b * SQ + srow) * DQ + h * HDQ + lane15;
    crow[0]  = (_Float16)(o0[r] * inv);
    crow[16] = (_Float16)(o1[r] * inv);
    crow[32] = (_Float16)(o2[r] * inv);
    crow[48] = (_Float16)(o3[r] * inv);
  }
}

// ---------------- host side -------------------------------------------------

extern "C" void kernel_launch(void* const* d_in, const int* in_sizes, int n_in,
                              void* d_out, int out_size, void* d_ws,
                              size_t ws_size, hipStream_t stream) {
  (void)in_sizes; (void)n_in; (void)out_size; (void)ws_size;

  const float* query = (const float*)d_in[0];
  const float* key   = (const float*)d_in[1];
  const float* value = (const float*)d_in[2];
  const float* mask  = (const float*)d_in[3];
  const float* Wq    = (const float*)d_in[4];
  const float* Wk    = (const float*)d_in[5];
  const float* Wv    = (const float*)d_in[6];
  const float* Wo    = (const float*)d_in[7];
  const float* bo    = (const float*)d_in[8];

  const size_t MB = 1ull << 20;
  char* ws = (char*)d_ws;
  _Float16* Xq  = (_Float16*)(ws + 0 * MB);    // 4096x1024 f16 (8 MiB)
  _Float16* Xk  = (_Float16*)(ws + 8 * MB);
  _Float16* Xv  = (_Float16*)(ws + 16 * MB);
  _Float16* Wqh = (_Float16*)(ws + 24 * MB);   // 1024x1024 f16 (2 MiB)
  _Float16* Wkh = (_Float16*)(ws + 26 * MB);
  _Float16* Wvh = (_Float16*)(ws + 28 * MB);
  _Float16* Woh = (_Float16*)(ws + 30 * MB);
  _Float16* Qh  = (_Float16*)(ws + 32 * MB);   // [B,H,S,HD] f16 (8 MiB)
  _Float16* Khh = (_Float16*)(ws + 40 * MB);   // [B,H,S,HD]
  _Float16* Vth = (_Float16*)(ws + 48 * MB);   // [B,H,HD,S]
  _Float16* Ctx = (_Float16*)(ws + 56 * MB);   // [B,S,D] f16

  const int nX = BQ * SQ * DQ;   // 4194304
  const int nW = DQ * DQ;        // 1048576

  cvt_f32_f16_kernel<<<nX / 4 / 256, 256, 0, stream>>>(query, Xq, nX / 4);
  cvt_f32_f16_kernel<<<nX / 4 / 256, 256, 0, stream>>>(key,   Xk, nX / 4);
  cvt_f32_f16_kernel<<<nX / 4 / 256, 256, 0, stream>>>(value, Xv, nX / 4);
  cvt_f32_f16_kernel<<<nW / 4 / 256, 256, 0, stream>>>(Wq, Wqh, nW / 4);
  cvt_f32_f16_kernel<<<nW / 4 / 256, 256, 0, stream>>>(Wk, Wkh, nW / 4);
  cvt_f32_f16_kernel<<<nW / 4 / 256, 256, 0, stream>>>(Wv, Wvh, nW / 4);
  cvt_f32_f16_kernel<<<nW / 4 / 256, 256, 0, stream>>>(Wo, Woh, nW / 4);

  dim3 gg(BQ * SQ / 64, DQ / 64);  // (64, 16)
  gemm_nt_kernel<<<gg, 128, 0, stream>>>(Xq, Wqh, Qh,  bo, DQ, 0, QSCALE);
  gemm_nt_kernel<<<gg, 128, 0, stream>>>(Xk, Wkh, Khh, bo, DQ, 0, 1.0f);
  gemm_nt_kernel<<<gg, 128, 0, stream>>>(Xv, Wvh, Vth, bo, DQ, 1, 1.0f);

  flash_attn_kernel<<<dim3(SQ / 64, HQ, BQ), 128, 0, stream>>>(Qh, Khh, Vth,
                                                               mask, Ctx);

  gemm_nt_kernel<<<gg, 128, 0, stream>>>(Ctx, Woh, d_out, bo, DQ, 2, 1.0f);
}
